// FlexibleGNN_841813590019
// MI455X (gfx1250) — matrix operand order repaired and checked by reading, not compiled, not run on hardware
//
#include <hip/hip_runtime.h>
#include <math.h>

typedef _Float16 v16h __attribute__((ext_vector_type(16)));
typedef float    v8f  __attribute__((ext_vector_type(8)));

#define LRELU_SLOPE 0.2f

// float atomic max via signed/unsigned integer atomics (works with -inf init)
__device__ __forceinline__ void atomic_max_float(float* addr, float val) {
    if (val >= 0.0f) {
        atomicMax((int*)addr, __float_as_int(val));
    } else {
        atomicMin((unsigned int*)addr, __float_as_uint(val));
    }
}

// ---------------------------------------------------------------------------
// Weight prep: convert f32 [L,64,64] (k-major) -> f16 transposed [L,64,64]
// (column-major per output channel) so B fragments are contiguous 16-half runs.
// ---------------------------------------------------------------------------
__global__ void prep_w(const float* __restrict__ Wl, const float* __restrict__ Wr,
                       _Float16* __restrict__ wlt, _Float16* __restrict__ wrt,
                       int total) {
    int i = blockIdx.x * blockDim.x + threadIdx.x;
    if (i >= total) return;               // total = L*64*64
    int l = i >> 12;
    int r = i & 4095;
    int k = r >> 6;
    int j = r & 63;
    wlt[(l << 12) + j * 64 + k] = (_Float16)Wl[i];
    wrt[(l << 12) + j * 64 + k] = (_Float16)Wr[i];
}

// ---------------------------------------------------------------------------
// Fused GEMM: xl = x@Wl + bl (wave 0), xr = x@Wr + br (wave 1).
// One 16-row node tile per block; per wave: 4 col-tiles x 2 K-steps of
// v_wmma_f32_16x16x32_f16 (f16 in, f32 accumulate).
// ---------------------------------------------------------------------------
__global__ void gat_gemm(const float* __restrict__ xin,
                         const _Float16* __restrict__ wlt,
                         const _Float16* __restrict__ wrt,
                         const float* __restrict__ bl,
                         const float* __restrict__ br,
                         float* __restrict__ xl,
                         float* __restrict__ xr,
                         int n) {
    const int wave = threadIdx.x >> 5;   // 0 -> lin_l, 1 -> lin_r
    const int lane = threadIdx.x & 31;
    const int g    = lane >> 4;          // half-wave group (ISA 16-bit layouts)
    const int l16  = lane & 15;
    const int tile = blockIdx.x;
    const bool full = (tile * 16 + 16) <= n;   // uniform per block

    int row = tile * 16 + l16;
    if (row >= n) row = n - 1;           // clamp: keep EXEC all-ones for WMMA
    const float2* xrow2 = (const float2*)(xin + (size_t)row * 64);

    // A fragments (16x32 f16): element s -> K = 16g + s (matches B)
    v16h a0, a1;
#pragma unroll
    for (int i = 0; i < 8; ++i) {
        const int bk = 16 * g + 2 * i;
        float2 p0 = xrow2[bk >> 1];
        float2 p1 = xrow2[(32 + bk) >> 1];
        a0[2 * i]     = (_Float16)p0.x;
        a0[2 * i + 1] = (_Float16)p0.y;
        a1[2 * i]     = (_Float16)p1.x;
        a1[2 * i + 1] = (_Float16)p1.y;
    }

    const _Float16* wt  = wave ? wrt : wlt;
    const float*    bv  = wave ? br  : bl;
    float*          out = wave ? xr  : xl;

#pragma unroll
    for (int ct = 0; ct < 4; ++ct) {
        const int col = ct * 16 + l16;
        // B fragment (32x16 f16): lanes 0-15 hold K=0..15, lanes 16-31 K=16..31
        // wlt region is 32B-aligned and offsets are multiples of 32B -> vector load
        const _Float16* wc = wt + col * 64 + g * 16;
        v16h b0 = *(const v16h*)wc;
        v16h b1 = *(const v16h*)(wc + 32);

        v8f c = {};
        c = __builtin_amdgcn_wmma_f32_16x16x32_f16(false, a0, false, b0,
                                                   (short)0, c, false, false);
        c = __builtin_amdgcn_wmma_f32_16x16x32_f16(false, a1, false, b1,
                                                   (short)0, c, false, false);

        const float bias  = bv[col];
        const int   rbase = tile * 16 + 8 * g;   // C layout: VGPR r -> M = r + 8g
        if (full) {
#pragma unroll
            for (int r = 0; r < 8; ++r)
                out[(size_t)(rbase + r) * 64 + col] = c[r] + bias;
        } else {
#pragma unroll
            for (int r = 0; r < 8; ++r) {
                const int orow = rbase + r;
                if (orow < n) out[(size_t)orow * 64 + col] = c[r] + bias;
            }
        }
    }
}

// ---------------------------------------------------------------------------
// Segment state + output accumulator init.
// ---------------------------------------------------------------------------
__global__ void init_seg(float* __restrict__ xout, float* __restrict__ mmax,
                         float* __restrict__ denom, int n) {
    int i = blockIdx.x * blockDim.x + threadIdx.x;
    if (i < n * 64) xout[i] = 0.0f;
    if (i < n) { mmax[i] = -INFINITY; denom[i] = 0.0f; }
}

// ---------------------------------------------------------------------------
// Per-edge attention logit: dot(leaky_relu(xl[src]+xr[dst]), att); seg-max.
// ---------------------------------------------------------------------------
__global__ void edge_logits(const float* __restrict__ xl, const float* __restrict__ xr,
                            const float* __restrict__ att,
                            const long long* __restrict__ src,
                            const long long* __restrict__ dst,
                            float* __restrict__ elog, float* __restrict__ mmax, int E) {
    int e = blockIdx.x * blockDim.x + threadIdx.x;
    if (e >= E) return;
    const long long s = src[e], t = dst[e];
    const float4* a4 = (const float4*)(xl + (size_t)s * 64);
    const float4* b4 = (const float4*)(xr + (size_t)t * 64);
    const float4* w4 = (const float4*)att;
    float acc = 0.0f;
#pragma unroll
    for (int i = 0; i < 16; ++i) {
        float4 a = a4[i], b = b4[i], w = w4[i];
        float v;
        v = a.x + b.x; v = v > 0.0f ? v : LRELU_SLOPE * v; acc += v * w.x;
        v = a.y + b.y; v = v > 0.0f ? v : LRELU_SLOPE * v; acc += v * w.y;
        v = a.z + b.z; v = v > 0.0f ? v : LRELU_SLOPE * v; acc += v * w.z;
        v = a.w + b.w; v = v > 0.0f ? v : LRELU_SLOPE * v; acc += v * w.w;
    }
    elog[e] = acc;
    atomic_max_float(&mmax[t], acc);
}

// ---------------------------------------------------------------------------
// exp(logit - max) and segment-sum of denominators.
// ---------------------------------------------------------------------------
__global__ void edge_exp(float* __restrict__ elog, const float* __restrict__ mmax,
                         const long long* __restrict__ dst,
                         float* __restrict__ denom, int E) {
    int e = blockIdx.x * blockDim.x + threadIdx.x;
    if (e >= E) return;
    const long long t = dst[e];
    float ex = expf(elog[e] - mmax[t]);
    elog[e] = ex;
    atomicAdd(&denom[t], ex);
}

// ---------------------------------------------------------------------------
// Weighted scatter: xout[dst] += alpha * xl[src]; 64 threads per edge.
// ---------------------------------------------------------------------------
__global__ void edge_scatter(const float* __restrict__ xl,
                             const float* __restrict__ elog,
                             const float* __restrict__ denom,
                             const long long* __restrict__ src,
                             const long long* __restrict__ dst,
                             float* __restrict__ xout, int E) {
    long long idx = (long long)blockIdx.x * blockDim.x + threadIdx.x;
    if (idx >= (long long)E * 64) return;
    int e = (int)(idx >> 6);
    int d = (int)(idx & 63);
    const long long s = src[e], t = dst[e];
    float alpha = elog[e] / denom[t];
    atomicAdd(&xout[(size_t)t * 64 + d], alpha * xl[(size_t)s * 64 + d]);
}

// ---------------------------------------------------------------------------
// Bias + exact-erf GELU, in place.
// ---------------------------------------------------------------------------
__global__ void finalize(float* __restrict__ xout, const float* __restrict__ bias,
                         int total) {
    int i = blockIdx.x * blockDim.x + threadIdx.x;
    if (i >= total) return;
    float v = xout[i] + bias[i & 63];
    xout[i] = 0.5f * v * (1.0f + erff(v * 0.70710678118654752f));
}

// ---------------------------------------------------------------------------
extern "C" void kernel_launch(void* const* d_in, const int* in_sizes, int n_in,
                              void* d_out, int out_size, void* d_ws, size_t ws_size,
                              hipStream_t stream) {
    const float*     x    = (const float*)d_in[0];
    const long long* ei   = (const long long*)d_in[1];   // int64 per reference
    const float*     Wl   = (const float*)d_in[2];
    const float*     bl   = (const float*)d_in[3];
    const float*     Wr   = (const float*)d_in[4];
    const float*     br   = (const float*)d_in[5];
    const float*     att  = (const float*)d_in[6];
    const float*     bias = (const float*)d_in[7];

    const int n  = in_sizes[0] / 64;          // 50000
    const int E  = in_sizes[1] / 2;           // 850000
    const int Lw = in_sizes[2] / 4096;        // 3

    // Workspace layout (floats)
    float* ws = (float*)d_ws;
    size_t o = 0;
    float* bufA  = ws + o; o += (size_t)n * 64;
    float* bufB  = ws + o; o += (size_t)n * 64;
    float* xlb   = ws + o; o += (size_t)n * 64;
    float* xrb   = ws + o; o += (size_t)n * 64;
    float* mmax  = ws + o; o += (size_t)n;
    float* denom = ws + o; o += (size_t)n;
    float* elog  = ws + o; o += (size_t)E;
    o = (o + 7) & ~(size_t)7;                 // 32B-align the f16 weight region
    _Float16* wlt = (_Float16*)(ws + o);
    _Float16* wrt = wlt + (size_t)Lw * 4096;

    const long long* src = ei;
    const long long* dst = ei + E;

    const int wtot = Lw * 4096;
    prep_w<<<(wtot + 255) / 256, 256, 0, stream>>>(Wl, Wr, wlt, wrt, wtot);

    const int tiles   = (n + 15) / 16;
    const int nb_node = (n * 64 + 255) / 256;
    const int nb_edge = (E + 255) / 256;
    const long long scat_tot = (long long)E * 64;
    const unsigned nb_scat = (unsigned)((scat_tot + 255) / 256);

    for (int l = 0; l < Lw; ++l) {
        const float* xin  = (l == 0) ? x : ((l & 1) ? bufA : bufB);
        float*       xout = (l == Lw - 1) ? (float*)d_out
                                          : ((l & 1) ? bufB : bufA);

        gat_gemm<<<tiles, 64, 0, stream>>>(xin, wlt + l * 4096, wrt + l * 4096,
                                           bl + l * 64, br + l * 64,
                                           xlb, xrb, n);
        init_seg<<<nb_node, 256, 0, stream>>>(xout, mmax, denom, n);
        edge_logits<<<nb_edge, 256, 0, stream>>>(xlb, xrb, att + l * 64,
                                                 src, dst, elog, mmax, E);
        edge_exp<<<nb_edge, 256, 0, stream>>>(elog, mmax, dst, denom, E);
        edge_scatter<<<nb_scat, 256, 0, stream>>>(xlb, elog, denom, src, dst,
                                                  xout, E);
        finalize<<<nb_node, 256, 0, stream>>>(xout, bias + l * 64, n * 64);
    }
}
